// BACENET_6201932775586
// MI455X (gfx1250) — compile-verified, matrix operand order checked by reading, not run on hardware
//
#include <hip/hip_runtime.h>
#include <hip/hip_bf16.h>
#include <math.h>

// ---------------- problem constants (from reference) ----------------
#define Nn     10000
#define Ee     320000
#define NRAD   8
#define NBb    8
#define RCc    5.0f
#define PIf    3.14159265358979323846f
#define NCOMP  48      // (ZETA+2)*NRAD
#define FEATD  192     // NCOMP*S
#define NCG    35      // 1+3+6+10+15 angular components

// workspace layout (floats)
#define WS_SEMB   0                       // N*4
#define WS_F0     (Nn*4)                  // N*32
#define WS_A      (Nn*4 + Nn*32)          // N*8*35*4
#define WS_TOTAL  (Nn*4 + Nn*32 + Nn*1120)

// multinomial term tables (a,b,c exponents for x,y,z), l-block id, coefficient
__constant__ unsigned char PA[NCG] = {0, 0,0,1, 0,0,0,1,1,2, 0,0,0,0,1,1,1,2,2,3,
                                      0,0,0,0,0,1,1,1,1,2,2,2,3,3,4};
__constant__ unsigned char PB[NCG] = {0, 0,1,0, 0,1,2,0,1,0, 0,1,2,3,0,1,2,0,1,0,
                                      0,1,2,3,4,0,1,2,3,0,1,2,0,1,0};
__constant__ unsigned char PCe[NCG]= {0, 1,0,0, 2,1,0,1,0,0, 3,2,1,0,2,1,0,1,0,0,
                                      4,3,2,1,0,3,2,1,0,2,1,0,1,0,0};
__constant__ unsigned char LBL[NCG]= {0, 1,1,1, 2,2,2,2,2,2, 3,3,3,3,3,3,3,3,3,3,
                                      4,4,4,4,4,4,4,4,4,4,4,4,4,4,4};
__constant__ float NMc[NCG] = {1.f, 1.f,1.f,1.f, 1.f,2.f,1.f,2.f,2.f,1.f,
                               1.f,3.f,3.f,1.f,3.f,6.f,3.f,3.f,3.f,1.f,
                               1.f,4.f,6.f,4.f,1.f,4.f,12.f,12.f,4.f,6.f,12.f,6.f,4.f,4.f,1.f};
__constant__ int CGSTART[5] = {0,1,4,10,20};
__constant__ int CGCOUNT[5] = {1,3,6,10,15};

typedef __attribute__((ext_vector_type(16))) _Float16 v16h;
typedef __attribute__((ext_vector_type(8)))  float    v8f;

// fast silu: x * v_rcp_f32(1 + exp(-x)) -- avoids the precise-divide sequence
__device__ __forceinline__ float silu(float x) {
  return x * __builtin_amdgcn_rcpf(1.f + __expf(-x));
}

// ---- WMMA fragment builders (ISA 7.12.2 16-bit layouts, wave32) ----
// A (16xK tile): lanes 0-15 -> M=lane, K halves {kb..kb+7, 16+kb..16+kb+7}, kb = 8*(lane>=16)
__device__ __forceinline__ v16h load_a_frag(const _Float16* sm, int row0, int ld,
                                            int k0, int kmax) {
  const int lane = threadIdx.x & 31;
  const int m  = row0 + (lane & 15);
  const int kb = (lane >> 4) << 3;
  v16h a;
#pragma unroll
  for (int j = 0; j < 8; ++j) {
    int k1 = k0 + kb + j;
    int k2 = k0 + 16 + kb + j;
    a[j]     = (k1 < kmax) ? sm[m * ld + k1] : (_Float16)0.f;
    a[8 + j] = (k2 < kmax) ? sm[m * ld + k2] : (_Float16)0.f;
  }
  return a;
}
// B (Kx16 tile) from LDS f16 row-major weights [K][ld]
__device__ __forceinline__ v16h load_b_frag(const _Float16* sm, int ld,
                                            int k0, int n0, int kmax) {
  const int lane = threadIdx.x & 31;
  const int n  = n0 + (lane & 15);
  const int kb = (lane >> 4) << 3;
  v16h b;
#pragma unroll
  for (int j = 0; j < 8; ++j) {
    int k1 = k0 + kb + j;
    int k2 = k0 + 16 + kb + j;
    b[j]     = (k1 < kmax) ? sm[k1 * ld + n] : (_Float16)0.f;
    b[8 + j] = (k2 < kmax) ? sm[k2 * ld + n] : (_Float16)0.f;
  }
  return b;
}
// B (Kx16 tile) from global f32 row-major weights [K][ld], converted to f16
__device__ __forceinline__ v16h load_b_frag_g(const float* W, int ld,
                                              int k0, int n0, int kmax) {
  const int lane = threadIdx.x & 31;
  const int n  = n0 + (lane & 15);
  const int kb = (lane >> 4) << 3;
  v16h b;
#pragma unroll
  for (int j = 0; j < 8; ++j) {
    int k1 = k0 + kb + j;
    int k2 = k0 + 16 + kb + j;
    b[j]     = (k1 < kmax) ? (_Float16)W[k1 * ld + n] : (_Float16)0.f;
    b[8 + j] = (k2 < kmax) ? (_Float16)W[k2 * ld + n] : (_Float16)0.f;
  }
  return b;
}

// ---------------- kernel 0: zero the accumulators ----------------
__global__ void k_zero(float* p, int n) {
  for (int i = blockIdx.x * blockDim.x + threadIdx.x; i < n;
       i += gridDim.x * blockDim.x)
    p[i] = 0.f;
}

// ---------------- kernel 1: species embedding (N x 4) ----------------
__global__ void k_semb(const int* __restrict__ species,
                       const float* __restrict__ Ws1, const float* __restrict__ bs1,
                       const float* __restrict__ Ws2, const float* __restrict__ bs2,
                       float* __restrict__ semb) {
  int n = blockIdx.x * blockDim.x + threadIdx.x;
  if (n >= Nn) return;
  int sp = species[n];
  float t[16];
#pragma unroll
  for (int j = 0; j < 16; ++j) t[j] = tanhf(Ws1[sp * 16 + j] + bs1[j]);
#pragma unroll
  for (int s = 0; s < 4; ++s) {
    float acc = bs2[s];
#pragma unroll
    for (int j = 0; j < 16; ++j) acc += t[j] * Ws2[j * 4 + s];
    semb[n * 4 + s] = acc;
  }
}

// ---------------- kernel 2: edge MLP (WMMA) + atomic scatter ----------------
#define EPB 128   // edges per block, 4 waves, 32 edges (2 M-tiles) per wave
__global__ void __launch_bounds__(EPB)
k_edge(const float* __restrict__ rij,
       const int* __restrict__ first, const int* __restrict__ second,
       const float* __restrict__ Wr1, const float* __restrict__ br1,
       const float* __restrict__ Wr2, const float* __restrict__ br2,
       const float* __restrict__ semb,
       float* __restrict__ feats0, float* __restrict__ Abuf) {
  __shared__ _Float16 besS[EPB * 8];      // 2 KB (K padded via kmax in frag load)
  __shared__ _Float16 h1S[EPB * 64];      // 16 KB hidden layer
  __shared__ float    Rbuf[EPB * 48];     // 24 KB radial features
  __shared__ float    fcS[EPB];
  __shared__ _Float16 wr1S[8 * 64];       // 1 KB
  __shared__ _Float16 wr2S[64 * 48];      // 6 KB
  __shared__ float    br1S[64];
  __shared__ float    br2S[48];

  const int tid = threadIdx.x;
  const int e   = blockIdx.x * EPB + tid;

  // ---- phase 0: stage weights into LDS (coalesced, once per block) ----
  for (int i = tid; i < 8 * 64; i += EPB)  wr1S[i] = (_Float16)Wr1[i];
  for (int i = tid; i < 64 * 48; i += EPB) wr2S[i] = (_Float16)Wr2[i];
  if (tid < 64) br1S[tid] = br1[tid];
  if (tid < 48) br2S[tid] = br2[tid];

  // ---- phase 1: per-edge scalar prep ----
  float rx = 0.f, ry = 0.f, rz = 0.f, fc = 0.f;
  if (e < Ee) {
    rx = rij[e * 3 + 0]; ry = rij[e * 3 + 1]; rz = rij[e * 3 + 2];
    float d  = sqrtf(rx * rx + ry * ry + rz * rz + 1e-12f);
    float iv = __builtin_amdgcn_rcpf(d);
    rx *= iv; ry *= iv; rz *= iv;
    fc = (d < RCc) ? 0.5f * (__cosf(PIf * d / RCc) + 1.f) : 0.f;
    float pref = sqrtf(2.f / RCc) * iv;
#pragma unroll
    for (int k = 1; k <= NBb; ++k)
      besS[tid * 8 + (k - 1)] = (_Float16)(pref * __sinf((float)k * PIf * d / RCc));
  } else {
#pragma unroll
    for (int k = 0; k < 8; ++k) besS[tid * 8 + k] = (_Float16)0.f;
  }
  fcS[tid] = fc;
  __syncthreads();

  // ---- phase 2: per-wave WMMA edge MLP (uniform control flow, EXEC all-ones) ----
  const int wave = tid >> 5;
  const int lane = tid & 31;
  const int nloc = lane & 15;
  const int mofs = (lane >> 4) << 3;

  for (int mt = 0; mt < 2; ++mt) {
    const int row0 = wave * 32 + mt * 16;

    // layer 1: bes(16x8 pad 32) @ Wr1(8x64) -> 16x64
    v16h a1 = load_a_frag(besS, row0, 8, 0, 8);
#pragma unroll
    for (int nt = 0; nt < 4; ++nt) {
      v16h b1 = load_b_frag(wr1S, 64, 0, nt * 16, 8);
      v8f z = {};
      v8f c = __builtin_amdgcn_wmma_f32_16x16x32_f16(false, a1, false, b1,
                                                     (short)0, z, false, false);
#pragma unroll
      for (int r = 0; r < 8; ++r) {
        int m = row0 + r + mofs;
        int n = nt * 16 + nloc;
        h1S[m * 64 + n] = (_Float16)silu(c[r] + br1S[n]);
      }
    }
    __syncthreads();

    // layer 2: h(16x64) @ Wr2(64x48) -> 16x48, bias+silu, *fc
#pragma unroll
    for (int nt = 0; nt < 3; ++nt) {
      v8f acc = {};
#pragma unroll
      for (int ks = 0; ks < 2; ++ks) {
        v16h a2 = load_a_frag(h1S, row0, 64, ks * 32, 64);
        v16h b2 = load_b_frag(wr2S, 48, ks * 32, nt * 16, 64);
        acc = __builtin_amdgcn_wmma_f32_16x16x32_f16(false, a2, false, b2,
                                                     (short)0, acc, false, false);
      }
#pragma unroll
      for (int r = 0; r < 8; ++r) {
        int m = row0 + r + mofs;
        int n = nt * 16 + nloc;
        Rbuf[m * 48 + n] = silu(acc[r] + br2S[n]) * fcS[m];
      }
    }
    __syncthreads();
  }

  // ---- phase 3: atomic segment-sum scatter (dominant cost; target fits in L2) ----
  if (e < Ee) {
    const int fa = first[e], sa = second[e];
    const float s0 = semb[sa * 4 + 0], s1 = semb[sa * 4 + 1];
    const float s2 = semb[sa * 4 + 2], s3 = semb[sa * 4 + 3];
    const float* Rrow = &Rbuf[tid * 48];

    // feats[0]: R[:, :8] x sj
    float* f0 = feats0 + (size_t)fa * 32;
#pragma unroll
    for (int r = 0; r < 8; ++r) {
      float Rr = Rrow[r];
      atomicAdd(f0 + r * 4 + 0, Rr * s0);
      atomicAdd(f0 + r * 4 + 1, Rr * s1);
      atomicAdd(f0 + r * 4 + 2, Rr * s2);
      atomicAdd(f0 + r * 4 + 3, Rr * s3);
    }

    // angular: Rl x comps x sj
    float px[5], py[5], pz[5];
    px[0] = py[0] = pz[0] = 1.f;
#pragma unroll
    for (int i = 1; i < 5; ++i) {
      px[i] = px[i - 1] * rx; py[i] = py[i - 1] * ry; pz[i] = pz[i - 1] * rz;
    }
    float* Ab = Abuf + (size_t)fa * 1120;  // [r][cg][s], s innermost (coalesced atomics)
#pragma unroll
    for (int cg = 0; cg < NCG; ++cg) {
      float cf = px[PA[cg]] * py[PB[cg]] * pz[PCe[cg]];
      int l = LBL[cg];
#pragma unroll
      for (int r = 0; r < 8; ++r) {
        float v = Rrow[8 + l * 8 + r] * cf;
        float* p = Ab + (r * NCG + cg) * 4;
        atomicAdd(p + 0, v * s0);
        atomicAdd(p + 1, v * s1);
        atomicAdd(p + 2, v * s2);
        atomicAdd(p + 3, v * s3);
      }
    }
  }
}

// ---------------- kernel 3: per-atom features + readout MLP (WMMA) ----------------
#define APB 64   // atoms per block, 4 waves x 16
__global__ void __launch_bounds__(128)
k_atom(const float* __restrict__ feats0, const float* __restrict__ Abuf,
       const float* __restrict__ Wa1, const float* __restrict__ ba1,
       const float* __restrict__ Wa2, const float* __restrict__ ba2,
       const float* __restrict__ Wa3, const float* __restrict__ ba3,
       float* __restrict__ out) {
  __shared__ _Float16 featS[APB * FEATD];  // 24 KB
  __shared__ _Float16 hS[APB * 64];        // 8 KB
  __shared__ _Float16 h2S[APB * 64];       // 8 KB
  __shared__ _Float16 wa2S[64 * 64];       // 8 KB

  const int tid = threadIdx.x;

  // stage Wa2 in LDS; prefetch Wa1 through the gfx1250 prefetch path
  for (int i = tid; i < 64 * 64; i += blockDim.x) wa2S[i] = (_Float16)Wa2[i];
  __builtin_prefetch(Wa1 + tid * 64, 0, 1);  // global_prefetch_b8, L2-resident weights

  // ---- build 192-dim feature vector: [feats0 | sum_c nm*A^2 per l-block] ----
  for (int idx = tid; idx < APB * FEATD; idx += blockDim.x) {
    int al = idx / FEATD, j = idx % FEATD;
    int atom = blockIdx.x * APB + al;
    float v = 0.f;
    if (atom < Nn) {
      int b = j >> 5;       // which 32-wide block (r*4+s within)
      int rs = j & 31;
      if (b == 0) {
        v = feats0[(size_t)atom * 32 + rs];
      } else {
        int l = b - 1, r = rs >> 2, s = rs & 3;
        const float* Ap = Abuf + ((size_t)atom * 8 + r) * (NCG * 4) + s;
        int c0 = CGSTART[l], cn = CGCOUNT[l];
        for (int c = 0; c < cn; ++c) {
          float av = Ap[(c0 + c) * 4];
          v += NMc[c0 + c] * av * av;
        }
      }
    }
    featS[al * FEATD + j] = (_Float16)v;
  }
  __syncthreads();

  const int wave = tid >> 5;
  const int lane = tid & 31;
  const int nloc = lane & 15;
  const int mofs = (lane >> 4) << 3;
  const int row0 = wave * 16;

  // layer 1: feat(16x192) @ Wa1(192x64)
#pragma unroll
  for (int nt = 0; nt < 4; ++nt) {
    v8f acc = {};
#pragma unroll
    for (int ks = 0; ks < 6; ++ks) {
      v16h a = load_a_frag(featS, row0, FEATD, ks * 32, FEATD);
      v16h b = load_b_frag_g(Wa1, 64, ks * 32, nt * 16, FEATD);
      acc = __builtin_amdgcn_wmma_f32_16x16x32_f16(false, a, false, b,
                                                   (short)0, acc, false, false);
    }
#pragma unroll
    for (int r = 0; r < 8; ++r) {
      int m = row0 + r + mofs;
      int n = nt * 16 + nloc;
      hS[m * 64 + n] = (_Float16)silu(acc[r] + ba1[n]);
    }
  }
  __syncthreads();

  // layer 2: h(16x64) @ Wa2(64x64)
#pragma unroll
  for (int nt = 0; nt < 4; ++nt) {
    v8f acc = {};
#pragma unroll
    for (int ks = 0; ks < 2; ++ks) {
      v16h a = load_a_frag(hS, row0, 64, ks * 32, 64);
      v16h b = load_b_frag(wa2S, 64, ks * 32, nt * 16, 64);
      acc = __builtin_amdgcn_wmma_f32_16x16x32_f16(false, a, false, b,
                                                   (short)0, acc, false, false);
    }
#pragma unroll
    for (int r = 0; r < 8; ++r) {
      int m = row0 + r + mofs;
      int n = nt * 16 + nloc;
      h2S[m * 64 + n] = (_Float16)silu(acc[r] + ba2[n]);
    }
  }
  __syncthreads();

  // layer 3: 64 -> 1 dot product
  for (int al = tid; al < APB; al += blockDim.x) {
    int atom = blockIdx.x * APB + al;
    if (atom < Nn) {
      float acc = ba3[0];
#pragma unroll
      for (int j = 0; j < 64; ++j) acc += (float)h2S[al * 64 + j] * Wa3[j];
      out[atom] = acc;
    }
  }
}

// ---------------- launcher ----------------
extern "C" void kernel_launch(void* const* d_in, const int* in_sizes, int n_in,
                              void* d_out, int out_size, void* d_ws, size_t ws_size,
                              hipStream_t stream) {
  const float* rij     = (const float*)d_in[0];
  const int*   species = (const int*)  d_in[1];
  const int*   first   = (const int*)  d_in[2];
  const int*   second  = (const int*)  d_in[3];
  const float* Wr1 = (const float*)d_in[4];
  const float* br1 = (const float*)d_in[5];
  const float* Wr2 = (const float*)d_in[6];
  const float* br2 = (const float*)d_in[7];
  const float* Ws1 = (const float*)d_in[8];
  const float* bs1 = (const float*)d_in[9];
  const float* Ws2 = (const float*)d_in[10];
  const float* bs2 = (const float*)d_in[11];
  const float* Wa1 = (const float*)d_in[12];
  const float* ba1 = (const float*)d_in[13];
  const float* Wa2 = (const float*)d_in[14];
  const float* ba2 = (const float*)d_in[15];
  const float* Wa3 = (const float*)d_in[16];
  const float* ba3 = (const float*)d_in[17];

  float* ws     = (float*)d_ws;
  float* semb   = ws + WS_SEMB;
  float* feats0 = ws + WS_F0;
  float* Abuf   = ws + WS_A;
  float* out    = (float*)d_out;

  // zero accumulators every call (deterministic: we re-accumulate fully)
  k_zero<<<2048, 256, 0, stream>>>(ws, WS_TOTAL);

  k_semb<<<(Nn + 255) / 256, 256, 0, stream>>>(species, Ws1, bs1, Ws2, bs2, semb);

  k_edge<<<(Ee + EPB - 1) / EPB, EPB, 0, stream>>>(rij, first, second,
                                                   Wr1, br1, Wr2, br2,
                                                   semb, feats0, Abuf);

  k_atom<<<(Nn + APB - 1) / APB, 128, 0, stream>>>(feats0, Abuf,
                                                   Wa1, ba1, Wa2, ba2, Wa3, ba3,
                                                   out);
}
